// TransLayer_69140383531228
// MI455X (gfx1250) — matrix operand order repaired and checked by reading, not compiled, not run on hardware
//
#include <hip/hip_runtime.h>
#include <hip/hip_bf16.h>

typedef __attribute__((ext_vector_type(16))) _Float16 v16h;
typedef __attribute__((ext_vector_type(8)))  _Float16 v8h;
typedef __attribute__((ext_vector_type(8)))  float    v8f;

// ---------------- constants ----------------
#define B_SZ   8
#define N_SEQ  4096
#define DIM_   512
#define HEADS_ 8
#define DH_    64
#define MLM    256          // landmarks
#define BH_    (B_SZ*HEADS_)
#define ND_    (B_SZ*N_SEQ) // 32768 tokens
#define LN_EPS_ 1e-5f

// =====================================================================
// Batched WMMA GEMM, NT layout only:
//   C(f32, MxN) = A(f16, MxK row-major) * B  with B stored transposed:
//   Bt(f16, NxK row-major), so all fragment loads are contiguous b128s.
// block = 128 threads (4 waves). Block tile = 128M x 64N.
// Each wave computes a 32M x 64N strip: 2 A-frags x 4 B-frags -> 8 WMMA
// per 32-wide K step. Requires M%128==0, N%64==0, K%32==0 (all true here).
// =====================================================================
__global__ void wmma_gemm_nt(const _Float16* __restrict__ A,
                             const _Float16* __restrict__ Bt,
                             float* __restrict__ C,
                             int Kdim, int lda, int ldb, int ldc,
                             long long sA, long long sB, long long sC)
{
    const int lane = threadIdx.x & 31;
    const int wave = threadIdx.x >> 5;      // 0..3
    const int half = lane >> 4;             // 0/1
    const int l16  = lane & 15;

    const int tileM = blockIdx.y * 128 + wave * 32;
    const int tileN = blockIdx.x * 64;
    const long long batch = blockIdx.z;

    const _Float16* Ab = A  + batch * sA;
    const _Float16* Bb = Bt + batch * sB;
    float*          Cb = C  + batch * sC;

    // row pointers (A: 2 sub-tiles; B: 4 sub-tiles), fragment offset baked in
    const _Float16* arow0 = Ab + (long long)(tileM +      l16) * lda + half * 16;
    const _Float16* arow1 = Ab + (long long)(tileM + 16 + l16) * lda + half * 16;
    const _Float16* brow0 = Bb + (long long)(tileN +      l16) * ldb + half * 16;
    const _Float16* brow1 = Bb + (long long)(tileN + 16 + l16) * ldb + half * 16;
    const _Float16* brow2 = Bb + (long long)(tileN + 32 + l16) * ldb + half * 16;
    const _Float16* brow3 = Bb + (long long)(tileN + 48 + l16) * ldb + half * 16;

    v8f acc[2][4];
#pragma unroll
    for (int i = 0; i < 2; ++i)
#pragma unroll
        for (int j = 0; j < 4; ++j) acc[i][j] = (v8f){};

    for (int kk = 0; kk < Kdim; kk += 32) {
        // prefetch the K-stream 512B (8 K-steps) ahead -> global_prefetch_b8
        __builtin_prefetch(arow0 + kk + 256, 0, 0);
        __builtin_prefetch(arow1 + kk + 256, 0, 0);
        __builtin_prefetch(brow0 + kk + 256, 0, 0);

        v16h af[2], bf[4];
        {   // A fragments (16x32, K contiguous per lane half: k = 16*half + s)
            v8h lo0 = *(const v8h*)(arow0 + kk);
            v8h hi0 = *(const v8h*)(arow0 + kk + 8);
            v8h lo1 = *(const v8h*)(arow1 + kk);
            v8h hi1 = *(const v8h*)(arow1 + kk + 8);
#pragma unroll
            for (int e = 0; e < 8; ++e) {
                af[0][e] = lo0[e]; af[0][8 + e] = hi0[e];
                af[1][e] = lo1[e]; af[1][8 + e] = hi1[e];
            }
        }
        {   // B fragments (32x16, K striped across lane halves; contiguous)
            const _Float16* bp[4] = { brow0 + kk, brow1 + kk, brow2 + kk, brow3 + kk };
#pragma unroll
            for (int j = 0; j < 4; ++j) {
                v8h b0 = *(const v8h*)(bp[j]);
                v8h b1 = *(const v8h*)(bp[j] + 8);
#pragma unroll
                for (int e = 0; e < 8; ++e) { bf[j][e] = b0[e]; bf[j][8 + e] = b1[e]; }
            }
        }
#pragma unroll
        for (int i = 0; i < 2; ++i)
#pragma unroll
            for (int j = 0; j < 4; ++j)
                acc[i][j] = __builtin_amdgcn_wmma_f32_16x16x32_f16(
                                false, af[i], false, bf[j], (short)0, acc[i][j],
                                false, false);
    }

    // store (documented 32-bit C/D layout: VGPR r -> rows r / r+8)
#pragma unroll
    for (int i = 0; i < 2; ++i)
#pragma unroll
        for (int j = 0; j < 4; ++j)
#pragma unroll
            for (int r = 0; r < 8; ++r)
                Cb[(long long)(tileM + i * 16 + r + half * 8) * ldc
                   + tileN + j * 16 + l16] = acc[i][j][r];
}

// =====================================================================
// Elementwise / reduction kernels (fp32 math)
// =====================================================================
__global__ void transpose_to_f16_kernel(const float* __restrict__ W,
                                        _Float16* __restrict__ WT,
                                        int rows, int cols)
{
    long long idx = (long long)blockIdx.x * 256 + threadIdx.x;
    if (idx >= (long long)rows * cols) return;
    int c = (int)(idx % cols);
    int r = (int)(idx / cols);
    WT[(long long)c * rows + r] = (_Float16)W[idx];
}

__global__ void layernorm_kernel(const float* __restrict__ x,
                                 const float* __restrict__ g,
                                 const float* __restrict__ bta,
                                 _Float16* __restrict__ h)
{
    const long long row = blockIdx.x;             // token
    const float* xr = x + row * DIM_;
    __shared__ float r1[256], r2[256];
    float s = 0.f, s2 = 0.f;
    for (int c = threadIdx.x; c < DIM_; c += 256) {
        float v = xr[c]; s += v; s2 += v * v;
    }
    r1[threadIdx.x] = s; r2[threadIdx.x] = s2; __syncthreads();
    for (int st = 128; st > 0; st >>= 1) {
        if (threadIdx.x < st) { r1[threadIdx.x] += r1[threadIdx.x + st];
                                r2[threadIdx.x] += r2[threadIdx.x + st]; }
        __syncthreads();
    }
    float mean = r1[0] * (1.0f / DIM_);
    float var  = r2[0] * (1.0f / DIM_) - mean * mean;
    float rstd = rsqrtf(var + LN_EPS_);
    for (int c = threadIdx.x; c < DIM_; c += 256) {
        float v = (xr[c] - mean) * rstd * g[c] + bta[c];
        h[row * DIM_ + c] = (_Float16)v;
    }
}

__global__ void qkv_split_kernel(const float* __restrict__ qkv,
                                 _Float16* __restrict__ q,
                                 _Float16* __restrict__ k,
                                 float* __restrict__ v,
                                 _Float16* __restrict__ vT)
{
    long long idx = (long long)blockIdx.x * 256 + threadIdx.x;
    if (idx >= (long long)ND_ * 3 * DIM_) return;
    int c        = (int)(idx % (3 * DIM_));
    long long t  = idx / (3 * DIM_);              // b*4096 + n
    int b  = (int)(t >> 12);
    int nI = (int)(t & 4095);
    int part = c / DIM_;
    int hh   = (c % DIM_) / DH_;
    int d    = c % DH_;
    long long bh = (long long)b * HEADS_ + hh;
    float val = qkv[idx];
    long long o = (bh * N_SEQ + nI) * DH_ + d;
    if (part == 0)       q[o] = (_Float16)(val * 0.125f);   // DH^-0.5
    else if (part == 1)  k[o] = (_Float16)val;
    else { v[o] = val; vT[(bh * DH_ + d) * N_SEQ + nI] = (_Float16)val; }
}

__global__ void pool_kernel(const _Float16* __restrict__ q,
                            const _Float16* __restrict__ k,
                            _Float16* __restrict__ ql,
                            _Float16* __restrict__ kl)
{
    long long idx = (long long)blockIdx.x * 256 + threadIdx.x;
    if (idx >= (long long)BH_ * MLM * DH_) return;
    int d        = (int)(idx % DH_);
    long long t  = idx / DH_;
    int m        = (int)(t % MLM);
    long long bh = t / MLM;
    long long base = (bh * N_SEQ + (long long)m * 16) * DH_ + d;
    float sq = 0.f, sk = 0.f;
#pragma unroll
    for (int i = 0; i < 16; ++i) {
        sq += (float)q[base + (long long)i * DH_];
        sk += (float)k[base + (long long)i * DH_];
    }
    ql[idx] = (_Float16)(sq * (1.0f / 16.0f));
    kl[idx] = (_Float16)(sk * (1.0f / 16.0f));
}

// one block per row; O32 may be null (or alias S in place)
__global__ void softmax_kernel(const float* __restrict__ S,
                               _Float16* __restrict__ O16,
                               float* __restrict__ O32,
                               int rowlen)
{
    long long row = blockIdx.x;
    const float* sr = S + row * (long long)rowlen;
    __shared__ float red[256];
    float m = -1e30f;
    for (int c = threadIdx.x; c < rowlen; c += 256) m = fmaxf(m, sr[c]);
    red[threadIdx.x] = m; __syncthreads();
    for (int st = 128; st > 0; st >>= 1) {
        if (threadIdx.x < st) red[threadIdx.x] = fmaxf(red[threadIdx.x], red[threadIdx.x + st]);
        __syncthreads();
    }
    m = red[0]; __syncthreads();
    float sum = 0.f;
    for (int c = threadIdx.x; c < rowlen; c += 256) sum += __expf(sr[c] - m);
    red[threadIdx.x] = sum; __syncthreads();
    for (int st = 128; st > 0; st >>= 1) {
        if (threadIdx.x < st) red[threadIdx.x] += red[threadIdx.x + st];
        __syncthreads();
    }
    float inv = 1.0f / red[0];
    for (int c = threadIdx.x; c < rowlen; c += 256) {
        float p = __expf(sr[c] - m) * inv;
        O16[row * (long long)rowlen + c] = (_Float16)p;
        if (O32) O32[row * (long long)rowlen + c] = p;
    }
}

__global__ void a2_reduce_kernel(const float* __restrict__ a2,
                                 float* __restrict__ rmax,
                                 float* __restrict__ cmax)
{
    int bh = blockIdx.x, j = threadIdx.x;
    const float* ap = a2 + (long long)bh * MLM * MLM;
    float cs = 0.f, rs = 0.f;
    for (int i = 0; i < MLM; ++i) {
        cs += fabsf(ap[(long long)i * MLM + j]);
        rs += fabsf(ap[(long long)j * MLM + i]);
    }
    __shared__ float r1[256], r2[256];
    r1[j] = rs; r2[j] = cs; __syncthreads();
    for (int st = 128; st > 0; st >>= 1) {
        if (j < st) { r1[j] = fmaxf(r1[j], r1[j + st]); r2[j] = fmaxf(r2[j], r2[j + st]); }
        __syncthreads();
    }
    if (j == 0) { rmax[bh] = r1[0]; cmax[bh] = r2[0]; }
}

__global__ void denom_kernel(const float* __restrict__ rmax,
                             const float* __restrict__ cmax,
                             float* __restrict__ denom)
{
    __shared__ float r[64], c[64];
    int t = threadIdx.x;
    r[t] = rmax[t]; c[t] = cmax[t]; __syncthreads();
    for (int st = 32; st > 0; st >>= 1) {
        if (t < st) { r[t] = fmaxf(r[t], r[t + st]); c[t] = fmaxf(c[t], c[t + st]); }
        __syncthreads();
    }
    if (t == 0) denom[0] = r[0] * c[0];
}

// z0 = a2^T / denom ; also z0^T = a2 / denom (B-operand layout)
__global__ void build_z0_kernel(const float* __restrict__ a2,
                                const float* __restrict__ denom,
                                _Float16* __restrict__ z,
                                _Float16* __restrict__ zT)
{
    long long idx = (long long)blockIdx.x * 256 + threadIdx.x;
    if (idx >= (long long)BH_ * MLM * MLM) return;
    int j = (int)(idx & 255);
    int i = (int)((idx >> 8) & 255);
    long long bh = idx >> 16;
    float inv = 1.0f / denom[0];
    z[idx]  = (_Float16)(a2[(bh << 16) + (long long)j * MLM + i] * inv);
    zT[idx] = (_Float16)(a2[idx] * inv);
}

// Et = (c1*I - c2*X)^T = c1*I - c2*X^T  (B-operand layout); optional Xf16 (A layout)
__global__ void diag_affine_T_kernel(const float* __restrict__ X,
                                     _Float16* __restrict__ Et,
                                     float c1, float c2,
                                     _Float16* __restrict__ Xh)
{
    long long idx = (long long)blockIdx.x * 256 + threadIdx.x;
    if (idx >= (long long)BH_ * MLM * MLM) return;
    int j = (int)(idx & 255);
    int i = (int)((idx >> 8) & 255);
    long long bh = idx >> 16;
    float xt = X[(bh << 16) + (long long)j * MLM + i];      // X^T element
    Et[idx] = (_Float16)(((i == j) ? c1 : 0.0f) - c2 * xt);
    if (Xh) Xh[idx] = (_Float16)X[idx];
}

// z = 0.25*P ; zT = 0.25*P^T
__global__ void convert_z_kernel(const float* __restrict__ P,
                                 _Float16* __restrict__ z,
                                 _Float16* __restrict__ zT)
{
    long long idx = (long long)blockIdx.x * 256 + threadIdx.x;
    if (idx >= (long long)BH_ * MLM * MLM) return;
    int j = (int)(idx & 255);
    int i = (int)((idx >> 8) & 255);
    long long bh = idx >> 16;
    z[idx]  = (_Float16)(P[idx] * 0.25f);
    zT[idx] = (_Float16)(P[(bh << 16) + (long long)j * MLM + i] * 0.25f);
}

// Gt[bh][d][m] = G[bh][m][d]  (f32 -> f16, B-operand layout for attn GEMM)
__global__ void convert_gt_kernel(const float* __restrict__ G,
                                  _Float16* __restrict__ Gt)
{
    long long idx = (long long)blockIdx.x * 256 + threadIdx.x;
    if (idx >= (long long)BH_ * DH_ * MLM) return;
    int m        = (int)(idx % MLM);
    long long t  = idx / MLM;
    int d        = (int)(t % DH_);
    long long bh = t / DH_;
    Gt[idx] = (_Float16)G[(bh * MLM + m) * DH_ + d];
}

__global__ void scale_to_f16_kernel(const float* __restrict__ in,
                                    _Float16* __restrict__ out,
                                    float s, long long count)
{
    long long idx = (long long)blockIdx.x * 256 + threadIdx.x;
    if (idx >= count) return;
    out[idx] = (_Float16)(in[idx] * s);
}

// o[token, h*64+d] = attn[b,h,n,d] + depthwise_conv33(v)[b,h,n,d]
__global__ void conv_merge_kernel(const float* __restrict__ attn,
                                  const float* __restrict__ v,
                                  const float* __restrict__ cw,
                                  _Float16* __restrict__ o)
{
    long long idx = (long long)blockIdx.x * 256 + threadIdx.x;
    if (idx >= (long long)BH_ * N_SEQ * DH_) return;
    int d        = (int)(idx & 63);
    long long t  = idx >> 6;
    int nI       = (int)(t & 4095);
    long long bh = t >> 12;
    int hh = (int)(bh & 7);
    int b  = (int)(bh >> 3);
    const float* vp = v + (bh * N_SEQ) * DH_ + d;
    float acc = attn[idx];
    for (int kk = 0; kk < 33; ++kk) {
        int src = nI + kk - 16;
        if (src >= 0 && src < N_SEQ)
            acc += cw[hh * 33 + kk] * vp[(long long)src * DH_];
    }
    long long token = (long long)b * N_SEQ + nI;
    o[token * DIM_ + hh * DH_ + d] = (_Float16)acc;
}

__global__ void final_kernel(const float* __restrict__ x,
                             const float* __restrict__ proj,
                             const float* __restrict__ bo,
                             float* __restrict__ y)
{
    long long idx = (long long)blockIdx.x * 256 + threadIdx.x;
    if (idx >= (long long)ND_ * DIM_) return;
    int c = (int)(idx % DIM_);
    y[idx] = x[idx] + proj[idx] + bo[c];
}

// =====================================================================
// Host side
// =====================================================================
extern "C" void kernel_launch(void* const* d_in, const int* in_sizes, int n_in,
                              void* d_out, int out_size, void* d_ws, size_t ws_size,
                              hipStream_t stream)
{
    (void)in_sizes; (void)n_in; (void)out_size; (void)ws_size;
    const float* x_in   = (const float*)d_in[0];
    const float* ln_g   = (const float*)d_in[1];
    const float* ln_b   = (const float*)d_in[2];
    const float* w_qkv  = (const float*)d_in[3];
    const float* w_out  = (const float*)d_in[4];
    const float* b_out  = (const float*)d_in[5];
    const float* conv_w = (const float*)d_in[6];
    float* y = (float*)d_out;

    // ---- workspace layout ----
    char* ws = (char*)d_ws;
    size_t off = 0;
    auto alloc = [&](size_t bytes) -> char* {
        char* p = ws + off;
        off += (bytes + 255) & ~(size_t)255;
        return p;
    };
    _Float16* wqkvT = (_Float16*)alloc((size_t)1536 * 512 * 2);
    _Float16* woutT = (_Float16*)alloc((size_t)512 * 512 * 2);
    _Float16* hbuf  = (_Float16*)alloc((size_t)ND_ * DIM_ * 2);
    float*    BIG   = (float*)   alloc((size_t)BH_ * N_SEQ * MLM * 4); // reused region
    _Float16* qf    = (_Float16*)alloc((size_t)BH_ * N_SEQ * DH_ * 2);
    _Float16* kf    = (_Float16*)alloc((size_t)BH_ * N_SEQ * DH_ * 2);
    float*    vf    = (float*)   alloc((size_t)BH_ * N_SEQ * DH_ * 4);
    _Float16* vT    = (_Float16*)alloc((size_t)BH_ * DH_ * N_SEQ * 2);
    _Float16* ql    = (_Float16*)alloc((size_t)BH_ * MLM * DH_ * 2);
    _Float16* kl    = (_Float16*)alloc((size_t)BH_ * MLM * DH_ * 2);
    _Float16* a1    = (_Float16*)alloc((size_t)BH_ * N_SEQ * MLM * 2);
    _Float16* a3    = (_Float16*)alloc((size_t)BH_ * MLM * N_SEQ * 2);
    float*    S2    = (float*)   alloc((size_t)BH_ * MLM * MLM * 4);
    _Float16* a2h   = (_Float16*)alloc((size_t)BH_ * MLM * MLM * 2);
    _Float16* zbuf  = (_Float16*)alloc((size_t)BH_ * MLM * MLM * 2);
    _Float16* zTbuf = (_Float16*)alloc((size_t)BH_ * MLM * MLM * 2);
    float*    Xf32  = (float*)   alloc((size_t)BH_ * MLM * MLM * 4);
    _Float16* Xf16  = (_Float16*)alloc((size_t)BH_ * MLM * MLM * 2);
    _Float16* Ebuf  = (_Float16*)alloc((size_t)BH_ * MLM * MLM * 2);
    float*    Pf32  = (float*)   alloc((size_t)BH_ * MLM * MLM * 4);
    float*    Gf32  = (float*)   alloc((size_t)BH_ * MLM * DH_ * 4);
    _Float16* Gt    = (_Float16*)alloc((size_t)BH_ * DH_ * MLM * 2);
    _Float16* H2h   = (_Float16*)alloc((size_t)BH_ * N_SEQ * MLM * 2);
    _Float16* obuf  = (_Float16*)alloc((size_t)ND_ * DIM_ * 2);
    float*    rmax  = (float*)   alloc(64 * 4);
    float*    cmax  = (float*)   alloc(64 * 4);
    float*    denom = (float*)   alloc(4);

    auto blocks = [](long long count) { return dim3((unsigned)((count + 255) / 256)); };

    // All GEMMs: NT layout (B stored N x K). M%128==0, N%64==0, K%32==0.
    auto gemm = [&](const _Float16* A, const _Float16* Bm, float* C,
                    int M, int N, int K, int lda, int ldb, int ldc,
                    long long sA, long long sB, long long sC, int batches) {
        dim3 grid(N / 64, M / 128, batches), block(128);
        wmma_gemm_nt<<<grid, block, 0, stream>>>(A, Bm, C, K, lda, ldb, ldc, sA, sB, sC);
    };

    // 1) weight transpose+convert (w_qkv: 512x1536, w_out: 512x512)
    transpose_to_f16_kernel<<<blocks(512LL * 1536), 256, 0, stream>>>(w_qkv, wqkvT, 512, 1536);
    transpose_to_f16_kernel<<<blocks(512LL * 512), 256, 0, stream>>>(w_out, woutT, 512, 512);

    // 2) LayerNorm
    layernorm_kernel<<<ND_, 256, 0, stream>>>(x_in, ln_g, ln_b, hbuf);

    // 3) qkv = h @ w_qkv -> BIG (f32, 32768x1536)
    gemm(hbuf, wqkvT, BIG, ND_, 1536, 512, 512, 512, 1536, 0, 0, 0, 1);

    // 4) split / scale / layouts
    qkv_split_kernel<<<blocks((long long)ND_ * 1536), 256, 0, stream>>>(BIG, qf, kf, vf, vT);

    // 5) landmark pooling
    pool_kernel<<<blocks((long long)BH_ * MLM * DH_), 256, 0, stream>>>(qf, kf, ql, kl);

    // 6) sim1 = q @ kl^T -> BIG ; softmax -> a1
    gemm(qf, kl, BIG, N_SEQ, MLM, DH_, DH_, DH_, MLM,
         (long long)N_SEQ * DH_, (long long)MLM * DH_, (long long)N_SEQ * MLM, BH_);
    softmax_kernel<<<(unsigned)(BH_ * N_SEQ), 256, 0, stream>>>(BIG, a1, nullptr, MLM);

    // 7) sim3 = ql @ k^T -> BIG ; softmax -> a3
    gemm(ql, kf, BIG, MLM, N_SEQ, DH_, DH_, DH_, N_SEQ,
         (long long)MLM * DH_, (long long)N_SEQ * DH_, (long long)MLM * N_SEQ, BH_);
    softmax_kernel<<<(unsigned)(BH_ * MLM), 256, 0, stream>>>(BIG, a3, nullptr, N_SEQ);

    // 8) sim2 = ql @ kl^T -> S2 ; softmax (keep f32 in place + f16)
    gemm(ql, kl, S2, MLM, MLM, DH_, DH_, DH_, MLM,
         (long long)MLM * DH_, (long long)MLM * DH_, (long long)MLM * MLM, BH_);
    softmax_kernel<<<(unsigned)(BH_ * MLM), 256, 0, stream>>>(S2, a2h, S2, MLM);

    // 9) pinv init: denom = max(rowsum)*max(colsum); z0 (+zT)
    a2_reduce_kernel<<<BH_, 256, 0, stream>>>(S2, rmax, cmax);
    denom_kernel<<<1, 64, 0, stream>>>(rmax, cmax, denom);
    build_z0_kernel<<<blocks((long long)BH_ * MLM * MLM), 256, 0, stream>>>(S2, denom, zbuf, zTbuf);

    // 10) Newton-Schulz: z = 0.25 z (13I - X(7I - X(15I - 7X))),  X = a2@z
    const long long cnt = (long long)BH_ * MLM * MLM;
    const long long sM = (long long)MLM * MLM;
    for (int it = 0; it < 6; ++it) {
        gemm(a2h, zTbuf, Xf32, MLM, MLM, MLM, MLM, MLM, MLM, sM, sM, sM, BH_);
        diag_affine_T_kernel<<<blocks(cnt), 256, 0, stream>>>(Xf32, Ebuf, 15.0f, 7.0f, Xf16);
        gemm(Xf16, Ebuf, Pf32, MLM, MLM, MLM, MLM, MLM, MLM, sM, sM, sM, BH_);
        diag_affine_T_kernel<<<blocks(cnt), 256, 0, stream>>>(Pf32, Ebuf, 7.0f, 1.0f, (_Float16*)nullptr);
        gemm(Xf16, Ebuf, Pf32, MLM, MLM, MLM, MLM, MLM, MLM, sM, sM, sM, BH_);
        diag_affine_T_kernel<<<blocks(cnt), 256, 0, stream>>>(Pf32, Ebuf, 13.0f, 1.0f, (_Float16*)nullptr);
        gemm(zbuf, Ebuf, Pf32, MLM, MLM, MLM, MLM, MLM, MLM, sM, sM, sM, BH_);
        convert_z_kernel<<<blocks(cnt), 256, 0, stream>>>(Pf32, zbuf, zTbuf);
    }

    // 11) G = a3 @ v  (NT against vT) -> Gf32 -> Gt (transposed f16)
    gemm(a3, vT, Gf32, MLM, DH_, N_SEQ, N_SEQ, N_SEQ, DH_,
         (long long)MLM * N_SEQ, (long long)DH_ * N_SEQ, (long long)MLM * DH_, BH_);
    convert_gt_kernel<<<blocks((long long)BH_ * DH_ * MLM), 256, 0, stream>>>(Gf32, Gt);

    // 12) H2 = a1 @ a2inv (B = zT) -> BIG (f32) -> H2h (f16)
    gemm(a1, zTbuf, BIG, N_SEQ, MLM, MLM, MLM, MLM, MLM,
         (long long)N_SEQ * MLM, sM, (long long)N_SEQ * MLM, BH_);
    scale_to_f16_kernel<<<blocks((long long)BH_ * N_SEQ * MLM), 256, 0, stream>>>(
        BIG, H2h, 1.0f, (long long)BH_ * N_SEQ * MLM);

    // 13) attn = H2 @ G (B = Gt) -> BIG (f32, b,h,n,dh)
    gemm(H2h, Gt, BIG, N_SEQ, DH_, MLM, MLM, MLM, DH_,
         (long long)N_SEQ * MLM, (long long)DH_ * MLM, (long long)N_SEQ * DH_, BH_);

    // 14) o = attn + depthwise conv residual, packed to (token, 512) f16
    conv_merge_kernel<<<blocks((long long)BH_ * N_SEQ * DH_), 256, 0, stream>>>(
        BIG, vf, conv_w, obuf);

    // 15) proj = o @ w_out (NT) -> BIG (f32, 32768x512)
    gemm(obuf, woutT, BIG, ND_, 512, 512, 512, 512, 512, 0, 0, 0, 1);

    // 16) y = x + proj + b_out
    final_kernel<<<blocks((long long)ND_ * DIM_), 256, 0, stream>>>(x_in, BIG, b_out, y);
}
